// RegionalSpatialAttention_71914932404728
// MI455X (gfx1250) — matrix operand order repaired and checked by reading, not compiled, not run on hardware
//
#include <hip/hip_runtime.h>
#include <hip/hip_bf16.h>

typedef __attribute__((ext_vector_type(16))) __bf16 v16bf;
typedef __attribute__((ext_vector_type(8)))  __bf16 bf16x8;
typedef __attribute__((ext_vector_type(8)))  float  v8f;

#define BATCH 4
#define HH 64
#define WW 64
#define CC 256
#define CK 256
#define HEADS 8
#define DD 32
#define SS 25
#define NPIX (BATCH * HH * WW)          // 16384
#define SCALE 0.17677669529663687f      // 1/sqrt(32)

// ---------------------------------------------------------------------------
// Prep: convert x -> bf16 row-major [NPIX, CC]; transpose+convert the four
// weight matrices W[k][n] -> WT[n][k] in bf16 so WMMA B-fragments are
// two contiguous 8-element K-chunks per lane.
// ---------------------------------------------------------------------------
__global__ void prep_kernel(const float* __restrict__ x,
                            const float* __restrict__ Wq,
                            const float* __restrict__ Wk,
                            const float* __restrict__ Wv,
                            const float* __restrict__ Wo,
                            __bf16* __restrict__ xb,
                            __bf16* __restrict__ WqT,
                            __bf16* __restrict__ WkT,
                            __bf16* __restrict__ WvT,
                            __bf16* __restrict__ WoT) {
  size_t i = (size_t)blockIdx.x * blockDim.x + threadIdx.x;
  if (i < (size_t)NPIX * CC) xb[i] = (__bf16)x[i];
  if (i < (size_t)CC * CK) {
    int k = (int)(i >> 8);
    int n = (int)(i & 255);
    size_t t = (size_t)n * CC + k;
    WqT[t] = (__bf16)Wq[i];
    WkT[t] = (__bf16)Wk[i];
    WvT[t] = (__bf16)Wv[i];
    WoT[t] = (__bf16)Wo[i];
  }
}

// ---------------------------------------------------------------------------
// bf16 WMMA GEMM: C[M=16384, N=256] = A[M, K=256] x B[K, N], B stored
// transposed (BT[n][k]).  One wave computes a 16 x (16*NT) strip for NMAT
// output matrices: each A fragment is loaded once and reused by NMAT*NT
// WMMAs (register-level reuse -> A L2 traffic cut by NT, plus 3x for the
// fused Q/K/V projection).
//
// Fragment layouts (ISA 7.12.2, wave32):
//   A (16x32 bf16): lane = M row (lanes 16-31 mirror, K offset +8); two
//     contiguous 8-elem chunks at k0+half*8 and k0+16+half*8.
//   B (32x16 bf16): lane = N col; two contiguous 8-elem chunks at k0+half*8 and k0+16+half*8.
//   C/D (16x16 f32): VGPR r holds M = r + 8*half, N = lane%16.
// ---------------------------------------------------------------------------
template <int NMAT, int NT>
__global__ void gemm_bf16_kernel(const __bf16* __restrict__ A,
                                 const __bf16* __restrict__ B0T,
                                 const __bf16* __restrict__ B1T,
                                 const __bf16* __restrict__ B2T,
                                 float* __restrict__ C0,
                                 float* __restrict__ C1,
                                 float* __restrict__ C2) {
  const int lane = threadIdx.x & 31;
  const int wave = threadIdx.x >> 5;
  const int nsb  = blockIdx.x;                 // N super-tile (NT 16-wide tiles)
  const int mb   = blockIdx.y * 4 + wave;      // M tile index
  const int half = lane >> 4;
  const int lr   = lane & 15;
  const int arow = mb * 16 + lr;

  const __bf16* Bt[3] = {B0T, B1T, B2T};
  float*        Cp[3] = {C0, C1, C2};

  v8f acc[NMAT * NT];
  #pragma unroll
  for (int i = 0; i < NMAT * NT; ++i) acc[i] = (v8f){};

  const __bf16* aBase = A + (size_t)arow * CC;

  #pragma unroll
  for (int k0 = 0; k0 < CC; k0 += 32) {
    // ---- A fragment: loaded once, feeds NMAT*NT WMMAs ----
    bf16x8 alo = *(const bf16x8*)(aBase + k0 + half * 8);
    bf16x8 ahi = *(const bf16x8*)(aBase + k0 + 16 + half * 8);
    v16bf a;
    #pragma unroll
    for (int i = 0; i < 8; ++i) { a[i] = alo[i]; a[i + 8] = ahi[i]; }

    #pragma unroll
    for (int m = 0; m < NMAT; ++m) {
      #pragma unroll
      for (int t = 0; t < NT; ++t) {
        const int bcol    = (nsb * NT + t) * 16 + lr;
        const size_t boff = (size_t)bcol * CC + k0 + half * 8;
        bf16x8 p = *(const bf16x8*)(Bt[m] + boff);
        bf16x8 q = *(const bf16x8*)(Bt[m] + boff + 16);
        v16bf b;
        #pragma unroll
        for (int i = 0; i < 8; ++i) { b[i] = p[i]; b[i + 8] = q[i]; }
        acc[m * NT + t] = __builtin_amdgcn_wmma_f32_16x16x32_bf16(
            false, a, false, b, (short)0, acc[m * NT + t], false, false);
      }
    }
  }

  // ---- store 16 x (16*NT) strip per output matrix ----
  #pragma unroll
  for (int m = 0; m < NMAT; ++m) {
    #pragma unroll
    for (int t = 0; t < NT; ++t) {
      #pragma unroll
      for (int r = 0; r < 8; ++r) {
        size_t o = (size_t)(mb * 16 + half * 8 + r) * CK + (nsb * NT + t) * 16 + lr;
        Cp[m][o] = acc[m * NT + t][r];
      }
    }
  }
}

// ---------------------------------------------------------------------------
// Attention core: one wave per (pixel, head); block = all 8 heads of one
// pixel.  Patch centers are clipped so neighbors never leave the image.
// Phase 1: lane = neighbor s (25 active), dot(Q, K_s + pos_s), wave softmax.
// Phase 2: lane = channel d, acc += w_s * V_s[d] with coalesced V reads.
// ---------------------------------------------------------------------------
__global__ void attn_kernel(const float* __restrict__ Q,
                            const float* __restrict__ K,
                            const float* __restrict__ V,
                            const float* __restrict__ pos,
                            __bf16* __restrict__ A2) {
  const int pix  = blockIdx.x;           // 0..16383
  const int head = threadIdx.x >> 5;     // 0..7
  const int lane = threadIdx.x & 31;

  const int b  = pix >> 12;
  const int hw = pix & 4095;
  const int h  = hw >> 6;
  const int w  = hw & 63;
  const int cy = min(max(h, 2), HH - 3);
  const int cx = min(max(w, 2), WW - 3);

  const float* qp = Q + (size_t)pix * CK + head * DD;

  // ---- phase 1: per-lane score for neighbor s = lane ----
  float score = -3.0e38f;
  {
    const int s  = lane;
    const int dy = s / 5 - 2;
    const int dx = s % 5 - 2;
    if (s < SS) {
      const int npix   = ((b * HH + cy + dy) * WW + (cx + dx));
      const float4* kp = (const float4*)(K + (size_t)npix * CK + head * DD);
      const float4* pp = (const float4*)(pos + (size_t)s * CK + head * DD);
      const float4* qv = (const float4*)qp;
      float acc = 0.f;
      #pragma unroll
      for (int i = 0; i < 8; ++i) {
        float4 qq = qv[i], kk = kp[i], pe = pp[i];
        acc += qq.x * (kk.x + pe.x);
        acc += qq.y * (kk.y + pe.y);
        acc += qq.z * (kk.z + pe.z);
        acc += qq.w * (kk.w + pe.w);
      }
      score = acc * SCALE;
    }
  }

  // ---- wave softmax over 25 lanes ----
  float m = score;
  #pragma unroll
  for (int off = 16; off > 0; off >>= 1) m = fmaxf(m, __shfl_xor(m, off, 32));
  float e = (lane < SS) ? __expf(score - m) : 0.f;
  float sum = e;
  #pragma unroll
  for (int off = 16; off > 0; off >>= 1) sum += __shfl_xor(sum, off, 32);
  const float wgt = e / sum;

  // ---- phase 2: lane = d, weighted V sum (coalesced 128B per s) ----
  float acc = 0.f;
  #pragma unroll
  for (int s = 0; s < SS; ++s) {
    const float ws = __shfl(wgt, s, 32);
    const int dy = s / 5 - 2;
    const int dx = s % 5 - 2;
    const int npix = ((b * HH + cy + dy) * WW + (cx + dx));
    acc += ws * V[(size_t)npix * CK + head * DD + lane];
  }

  A2[(size_t)pix * CK + head * DD + lane] = (__bf16)acc;
}

// ---------------------------------------------------------------------------
extern "C" void kernel_launch(void* const* d_in, const int* in_sizes, int n_in,
                              void* d_out, int out_size, void* d_ws, size_t ws_size,
                              hipStream_t stream) {
  const float* x   = (const float*)d_in[0];
  const float* Wq  = (const float*)d_in[1];
  const float* Wk  = (const float*)d_in[2];
  const float* Wv  = (const float*)d_in[3];
  const float* Wo  = (const float*)d_in[4];
  const float* pos = (const float*)d_in[5];
  float* out = (float*)d_out;

  // workspace layout
  char* p = (char*)d_ws;
  float* Qf = (float*)p;  p += (size_t)NPIX * CK * sizeof(float);
  float* Kf = (float*)p;  p += (size_t)NPIX * CK * sizeof(float);
  float* Vf = (float*)p;  p += (size_t)NPIX * CK * sizeof(float);
  __bf16* xb  = (__bf16*)p; p += (size_t)NPIX * CC * sizeof(__bf16);
  __bf16* A2  = (__bf16*)p; p += (size_t)NPIX * CK * sizeof(__bf16);
  __bf16* WqT = (__bf16*)p; p += (size_t)CC * CK * sizeof(__bf16);
  __bf16* WkT = (__bf16*)p; p += (size_t)CC * CK * sizeof(__bf16);
  __bf16* WvT = (__bf16*)p; p += (size_t)CC * CK * sizeof(__bf16);
  __bf16* WoT = (__bf16*)p; p += (size_t)CC * CK * sizeof(__bf16);

  // 1) precision/layout prep
  prep_kernel<<<(NPIX * CC) / 256, 256, 0, stream>>>(x, Wq, Wk, Wv, Wo,
                                                     xb, WqT, WkT, WvT, WoT);

  // 2) fused Q/K/V projection GEMMs: NMAT=3, NT=2 -> 6 WMMAs per A fragment
  dim3 pgrid(CK / (16 * 2), NPIX / (16 * 4));
  gemm_bf16_kernel<3, 2><<<pgrid, 128, 0, stream>>>(xb, WqT, WkT, WvT, Qf, Kf, Vf);

  // 3) attention core (softmax over 25 neighbors, 8 heads per block)
  attn_kernel<<<NPIX, 256, 0, stream>>>(Qf, Kf, Vf, pos, A2);

  // 4) output projection GEMM: NMAT=1, NT=4 -> 4 WMMAs per A fragment
  dim3 ogrid(CK / (16 * 4), NPIX / (16 * 4));
  gemm_bf16_kernel<1, 4><<<ogrid, 128, 0, stream>>>(A2, WoT, WoT, WoT, out, out, out);
}
